// DLA_38225208934978
// MI455X (gfx1250) — matrix operand order repaired and checked by reading, not compiled, not run on hardware
//
#include <hip/hip_runtime.h>
#include <math.h>

// ---------------------------------------------------------------------------
// Problem constants (match reference)
// ---------------------------------------------------------------------------
#define SEQ   2048
#define DIM   2048
#define NH    16
#define QLR   768
#define KVLR  512
#define DN    128
#define DR    64
#define DV    128
#define QKH   192          // DN + DR
#define HHALF 96           // QKH / 2

#define SCALE_C       0.07216878364870323f  // QKH^-0.5
#define LAMBDA_INIT_C 0.5560582043409389f   // 0.8 - 0.6*exp(-0.9)
#define OUT_SCALE_C   0.4439417956590611f   // 1 - LAMBDA_INIT
#define MAX_SCORE_C   100.0f
#define NEG_BIG      -1e9f

// ---------------------------------------------------------------------------
// CDNA5 WMMA types
// ---------------------------------------------------------------------------
typedef __attribute__((ext_vector_type(16))) __bf16 v16bf;
typedef __attribute__((ext_vector_type(8)))  float  v8f;

static __device__ __forceinline__ v8f wmma_bf16(v16bf a, v16bf b, v8f c) {
    // D = A(16x32 bf16) * B(32x16 bf16) + C(16x16 f32)
    return __builtin_amdgcn_wmma_f32_16x16x32_bf16(
        false, a, false, b, (short)0, c, false, false);
}

#define V8F_ZERO {0.f,0.f,0.f,0.f,0.f,0.f,0.f,0.f}

union FragU { uint4 u[2]; v16bf v; };

// ---------------------------------------------------------------------------
// Fragment loaders (bf16 source; global or LDS).
// A 16x32: lane m(0..15) row m: halves[0..7]=K d..d+7, halves[8..15]=K d+16..23
//          lane m+16    row m: halves = K d+8..d+15, K d+24..d+31
// B 32x16: lane n(0..15) col n: halves = K d..d+7, K d+16..d+23 ; lane n+16: K d+8..d+15, K d+24..d+31
// ---------------------------------------------------------------------------
static __device__ __forceinline__ v16bf load_a_bf16(const __bf16* __restrict__ A,
                                                    int lda, int m0, int k) {
    const int lane = threadIdx.x & 31, hi = lane >> 4, lm = lane & 15;
    const __bf16* p = A + (size_t)(m0 + lm) * lda + k + hi * 8;
    FragU x;
    x.u[0] = *reinterpret_cast<const uint4*>(p);
    x.u[1] = *reinterpret_cast<const uint4*>(p + 16);
    return x.v;
}

static __device__ __forceinline__ v16bf load_b_bf16(const __bf16* __restrict__ B,
                                                    int ldb, int n0, int k) {
    const int lane = threadIdx.x & 31, hi = lane >> 4, lm = lane & 15;
    const __bf16* p = B + (size_t)(n0 + lm) * ldb + k + hi * 8;
    FragU x;
    x.u[0] = *reinterpret_cast<const uint4*>(p);
    x.u[1] = *reinterpret_cast<const uint4*>(p + 16);
    return x.v;
}

// p fragment (full K=32) from LDS f32 tile of 32 score columns, row stride 33:
// cols 0..15 = key subtile 0, cols 16..31 = key subtile 1.
static __device__ __forceinline__ v16bf pfrag_from_lds32(const float* __restrict__ lds) {
    const int lane = threadIdx.x & 31, hi = lane >> 4, lm = lane & 15;
    const float* row = lds + lm * 33 + hi * 8;
    v16bf a;
#pragma unroll
    for (int i = 0; i < 8; ++i) a[i] = (__bf16)row[i];        // K d..d+7 (+8 for hi)
#pragma unroll
    for (int i = 0; i < 8; ++i) a[8 + i] = (__bf16)row[16 + i]; // K d+16..d+23 (+8)
    return a;
}

// V fragment: full 32-deep B column (dv) read from dv-major vt[h][dv][t].
static __device__ __forceinline__ v16bf load_v32(const __bf16* __restrict__ vth,
                                                 int dv0, int jb) {
    const int lane = threadIdx.x & 31, hi = lane >> 4, lm = lane & 15;
    const __bf16* p = vth + (size_t)(dv0 + lm) * SEQ + jb + hi * 8;
    FragU x;
    x.u[0] = *reinterpret_cast<const uint4*>(p);
    x.u[1] = *reinterpret_cast<const uint4*>(p + 16);
    return x.v;
}

// ---------------------------------------------------------------------------
// CDNA5 async global->LDS copy (ASYNCcnt-tracked), §15.18 VGLOBAL op 98.
// ---------------------------------------------------------------------------
static __device__ __forceinline__ unsigned lds_off(const void* p) {
    return (unsigned)(unsigned long long)p;   // low 32 bits = LDS byte offset
}
static __device__ __forceinline__ void async_copy_b128(unsigned lds_byte_off,
                                                       const void* gaddr) {
    asm volatile("global_load_async_to_lds_b128 %0, %1, off"
                 :: "v"(lds_byte_off), "v"((unsigned long long)gaddr)
                 : "memory");
}
static __device__ __forceinline__ void wait_async0() {
    asm volatile("s_wait_asynccnt 0x0" ::: "memory");
}

// ---------------------------------------------------------------------------
// GEMM: C[M,N] = A[M,K] * B[N,K]^T, bf16 in, bf16 or f32 out, f32 accumulate.
// 128 threads = 4 waves; wave tile 64(M) x 32(N); block tile 128 x 64.
// ---------------------------------------------------------------------------
template <bool OUT_BF16>
__global__ __launch_bounds__(128) void gemm_bt_kernel(
    const __bf16* __restrict__ A, int lda,
    const __bf16* __restrict__ B, int ldb,
    void* __restrict__ Cp, int ldc, int K)
{
    const int lane = threadIdx.x & 31, wave = threadIdx.x >> 5;
    const int hi = lane >> 4, lm = lane & 15;
    const int m0 = blockIdx.y * 128 + (wave & 1) * 64;
    const int n0 = blockIdx.x * 64 + (wave >> 1) * 32;

    v8f acc[4][2];
#pragma unroll
    for (int i = 0; i < 4; ++i) {
        v8f z = V8F_ZERO;
        acc[i][0] = z; acc[i][1] = z;
    }

    for (int k = 0; k < K; k += 32) {
        if (k + 32 < K) {
            __builtin_prefetch(A + (size_t)(m0 + lm) * lda + k + 32, 0, 1);
            __builtin_prefetch(B + (size_t)(n0 + lm) * ldb + k + 32, 0, 1);
        }
        v16bf b0 = load_b_bf16(B, ldb, n0, k);
        v16bf b1 = load_b_bf16(B, ldb, n0 + 16, k);
#pragma unroll
        for (int i = 0; i < 4; ++i) {
            v16bf a = load_a_bf16(A, lda, m0 + 16 * i, k);
            acc[i][0] = wmma_bf16(a, b0, acc[i][0]);
            acc[i][1] = wmma_bf16(a, b1, acc[i][1]);
        }
    }

#pragma unroll
    for (int i = 0; i < 4; ++i) {
#pragma unroll
        for (int r = 0; r < 8; ++r) {
            const size_t row = (size_t)(m0 + 16 * i + r + 8 * hi) * ldc;
            if (OUT_BF16) {
                __bf16* C = (__bf16*)Cp;
                C[row + n0 + lm]      = (__bf16)acc[i][0][r];
                C[row + n0 + 16 + lm] = (__bf16)acc[i][1][r];
            } else {
                float* C = (float*)Cp;
                C[row + n0 + lm]      = acc[i][0][r];
                C[row + n0 + 16 + lm] = acc[i][1][r];
            }
        }
    }
}

// ---------------------------------------------------------------------------
// f32 -> bf16 bulk convert (n multiple of 4)
// ---------------------------------------------------------------------------
__global__ __launch_bounds__(256) void cvt_bf16_kernel(const float* __restrict__ in,
                                                       __bf16* __restrict__ out,
                                                       size_t n)
{
    size_t i = ((size_t)blockIdx.x * blockDim.x + threadIdx.x) * 4;
    if (i >= n) return;
    float4 v = *reinterpret_cast<const float4*>(in + i);
    out[i]     = (__bf16)v.x;
    out[i + 1] = (__bf16)v.y;
    out[i + 2] = (__bf16)v.z;
    out[i + 3] = (__bf16)v.w;
}

// ---------------------------------------------------------------------------
// lambda = exp(dot(lqn,lkn)) - exp(dot(lqr,lkr)) + LAMBDA_INIT  (one wave)
// ---------------------------------------------------------------------------
__global__ __launch_bounds__(32) void lam_kernel(
    const float* __restrict__ lqn, const float* __restrict__ lqr,
    const float* __restrict__ lkn, const float* __restrict__ lkr,
    float* __restrict__ lam_out)
{
    const int lane = threadIdx.x;
    float d1 = 0.f, d2 = 0.f;
    for (int i = lane; i < DN; i += 32) d1 += lqn[i] * lkn[i];
    for (int i = lane; i < DR; i += 32) d2 += lqr[i] * lkr[i];
#pragma unroll
    for (int s = 16; s > 0; s >>= 1) {
        d1 += __shfl_xor(d1, s, 32);
        d2 += __shfl_xor(d2, s, 32);
    }
    if (lane == 0) *lam_out = __expf(d1) - __expf(d2) + LAMBDA_INIT_C;
}

// ---------------------------------------------------------------------------
// Prep (all bf16):
//   qbuf[t][h][128..191] <- rope(q_pe) in place   (qbuf doubles as qf)
//   kf[t][h][0..191]     <- [kvb k_nope | rope(k_pe) bcast over h]
//   vt[h][dv][t]         <- kvb v (dv-major)
// ---------------------------------------------------------------------------
__global__ __launch_bounds__(160) void prep_kernel(
    __bf16* __restrict__ qbuf,          // [SEQ, NH*QKH] (rope in place)
    const __bf16* __restrict__ kvb,     // [SEQ, NH*(DN+DV)]
    const __bf16* __restrict__ kvfull,  // [SEQ, KVLR+DR]
    const float* __restrict__ fcos,     // [SEQ, DR/2]
    const float* __restrict__ fsin,
    __bf16* __restrict__ kf, __bf16* __restrict__ vt)
{
    const int t = blockIdx.x, h = blockIdx.y, tid = threadIdx.x;
    const __bf16* kr = kvb + (size_t)t * (NH * (DN + DV)) + h * (DN + DV);
    __bf16* qfr = qbuf + ((size_t)t * NH + h) * QKH;
    __bf16* kfr = kf   + ((size_t)t * NH + h) * QKH;

    if (tid < 128) {
        kfr[tid] = kr[tid];                                   // k_nope
        vt[(size_t)(h * DV + tid) * SEQ + t] = kr[DN + tid];  // v transposed
    } else {
        const int i = tid - 128;                              // rope pair 0..31
        const float c  = fcos[(size_t)t * (DR / 2) + i];
        const float sn = fsin[(size_t)t * (DR / 2) + i];
        const float xr = (float)qfr[DN + 2 * i], xi = (float)qfr[DN + 2 * i + 1];
        qfr[DN + 2 * i]     = (__bf16)(xr * c - xi * sn);
        qfr[DN + 2 * i + 1] = (__bf16)(xr * sn + xi * c);
        const float kxr = (float)kvfull[(size_t)t * (KVLR + DR) + KVLR + 2 * i];
        const float kxi = (float)kvfull[(size_t)t * (KVLR + DR) + KVLR + 2 * i + 1];
        kfr[DN + 2 * i]     = (__bf16)(kxr * c - kxi * sn);
        kfr[DN + 2 * i + 1] = (__bf16)(kxr * sn + kxi * c);
    }
}

// ---------------------------------------------------------------------------
// Async prefetch of a 32-key K panel (32 rows x 192 dims bf16 = 12 KB):
// 768 x 16B chunks, 24 async b128 per wave.
// ---------------------------------------------------------------------------
static __device__ __forceinline__ void prefetch_k32(const __bf16* __restrict__ kfh,
                                                    int ld, int jb,
                                                    __bf16* __restrict__ stage) {
    const int lane = threadIdx.x & 31;
#pragma unroll
    for (int i = 0; i < 24; ++i) {
        const int c = i * 32 + lane;
        const int row = c / 24, col8 = c % 24;   // 24 chunks per 384B row
        async_copy_b128(lds_off(stage + row * QKH + col8 * 8),
                        kfh + (size_t)(jb + row) * ld + col8 * 8);
    }
}

// ---------------------------------------------------------------------------
// Flash-style differential attention, one wave per (16-row q tile, head).
// 32 keys per step: two score subtiles -> one full-K p fragment, so every
// p@V WMMA uses all 32 K slots. K panel async-double-buffered in LDS.
// Grid (SEQ/16, NH), block 32.
// ---------------------------------------------------------------------------
__global__ __launch_bounds__(32) void diff_attn_kernel(
    const __bf16* __restrict__ qf,   // [SEQ, NH, QKH] (rope applied)
    const __bf16* __restrict__ kf,   // [SEQ, NH, QKH]
    const __bf16* __restrict__ vt,   // [NH, DV, SEQ]
    const float*  __restrict__ lam_p,
    __bf16* __restrict__ out)        // [SEQ, NH*DV]
{
    __shared__ __bf16 kstage[2][32 * QKH];   // 2 x 12 KB
    __shared__ float  lds_s1[16 * 33];       // 32 score cols (2 subtiles), pad 1
    __shared__ float  lds_s2[16 * 33];
    __shared__ float  lds_f[32];

    const int jq = blockIdx.x, h = blockIdx.y;
    const int lane = threadIdx.x & 31, hi = lane >> 4, lm = lane & 15;
    const int q0 = jq * 16;
    const int ld = NH * QKH;                 // 3072
    const float lam = *lam_p;
    const __bf16* kfh = kf + h * QKH;
    const __bf16* vth = vt + (size_t)h * DV * SEQ;

    // q fragments (A layout), loaded once
    v16bf qa1[3], qa2[3];
#pragma unroll
    for (int dk = 0; dk < 3; ++dk) {
        qa1[dk] = load_a_bf16(qf + h * QKH,         ld, q0, dk * 32);
        qa2[dk] = load_a_bf16(qf + h * QKH + HHALF, ld, q0, dk * 32);
    }

    v8f acc1[8], acc2[8];
#pragma unroll
    for (int nt = 0; nt < 8; ++nt) {
        v8f z = V8F_ZERO;
        acc1[nt] = z; acc2[nt] = z;
    }
    float m1 = -1e30f, l1 = 0.f, m2 = -1e30f, l2 = 0.f;

    prefetch_k32(kfh, ld, 0, kstage[0]);     // pipeline prologue

    for (int jb = 0; jb <= q0; jb += 32) {
        wait_async0();                       // panel [jb, jb+32) resident
        if (jb + 32 <= q0)                   // overlap next panel with compute
            prefetch_k32(kfh, ld, jb + 32, kstage[((jb >> 5) + 1) & 1]);
        const __bf16* ks = kstage[(jb >> 5) & 1];

        // Two 16-key score subtiles per step (causal: subtile 1 on the
        // diagonal step is entirely masked -> fill with NEG_BIG).
#pragma unroll
        for (int si = 0; si < 2; ++si) {
            const int kb = jb + si * 16;
            if (kb <= q0) {
                v8f s1 = V8F_ZERO, s2 = V8F_ZERO;
#pragma unroll
                for (int dk = 0; dk < 3; ++dk)
                    s1 = wmma_bf16(qa1[dk],
                                   load_b_bf16(ks + (size_t)si * 16 * QKH, QKH, 0, dk * 32), s1);
#pragma unroll
                for (int dk = 0; dk < 3; ++dk)
                    s2 = wmma_bf16(qa2[dk],
                                   load_b_bf16(ks + (size_t)si * 16 * QKH + HHALF, QKH, 0, dk * 32), s2);
#pragma unroll
                for (int r = 0; r < 8; ++r) {
                    const int row = q0 + r + 8 * hi;
                    const int col = kb + lm;
                    float v1 = fminf(fmaxf(s1[r] * SCALE_C, -MAX_SCORE_C), MAX_SCORE_C);
                    float v2 = fminf(fmaxf(s2[r] * SCALE_C, -MAX_SCORE_C), MAX_SCORE_C);
                    if (col > row) { v1 += NEG_BIG; v2 += NEG_BIG; }
                    lds_s1[(r + 8 * hi) * 33 + si * 16 + lm] = v1;
                    lds_s2[(r + 8 * hi) * 33 + si * 16 + lm] = v2;
                }
            } else {
#pragma unroll
                for (int r = 0; r < 8; ++r) {
                    lds_s1[(r + 8 * hi) * 33 + si * 16 + lm] = NEG_BIG;
                    lds_s2[(r + 8 * hi) * 33 + si * 16 + lm] = NEG_BIG;
                }
            }
        }
        __syncthreads();

        // dual online-softmax row update over 32 columns (lane L owns row L)
        if (lane < 16) {
            float* r1 = &lds_s1[lane * 33];
            float* r2 = &lds_s2[lane * 33];
            float rm1 = -1e30f, rm2 = -1e30f;
#pragma unroll
            for (int c = 0; c < 32; ++c) { rm1 = fmaxf(rm1, r1[c]); rm2 = fmaxf(rm2, r2[c]); }
            const float nm1 = fmaxf(m1, rm1), nm2 = fmaxf(m2, rm2);
            const float f1 = __expf(m1 - nm1), f2 = __expf(m2 - nm2);
            float sum1 = 0.f, sum2 = 0.f;
#pragma unroll
            for (int c = 0; c < 32; ++c) {
                const float p1 = __expf(r1[c] - nm1); r1[c] = p1; sum1 += p1;
                const float p2 = __expf(r2[c] - nm2); r2[c] = p2; sum2 += p2;
            }
            l1 = l1 * f1 + sum1; m1 = nm1;
            l2 = l2 * f2 + sum2; m2 = nm2;
            lds_f[lane] = f1; lds_f[16 + lane] = f2;
        }
        __syncthreads();

        // rescale accumulators once per 32 keys; p @ V with full K=32 frags
        float f1s[8], f2s[8];
#pragma unroll
        for (int r = 0; r < 8; ++r) {
            f1s[r] = lds_f[r + 8 * hi];
            f2s[r] = lds_f[16 + r + 8 * hi];
        }
        v16bf pa1 = pfrag_from_lds32(lds_s1);
        v16bf pa2 = pfrag_from_lds32(lds_s2);
#pragma unroll
        for (int nt = 0; nt < 8; ++nt) {
#pragma unroll
            for (int r = 0; r < 8; ++r) { acc1[nt][r] *= f1s[r]; acc2[nt][r] *= f2s[r]; }
            v16bf vf = load_v32(vth, nt * 16, jb);  // masked keys hit p==0
            acc1[nt] = wmma_bf16(pa1, vf, acc1[nt]);
            acc2[nt] = wmma_bf16(pa2, vf, acc2[nt]);
        }
        __syncthreads();
    }

    // epilogue: (acc1/l1 - lam*acc2/l2) * (1 - LAMBDA_INIT), store bf16
    if (lane < 16) {
        lds_f[lane]      = 1.f / l1;
        lds_f[16 + lane] = 1.f / l2;
    }
    __syncthreads();
    float il1[8], il2[8];
#pragma unroll
    for (int r = 0; r < 8; ++r) {
        il1[r] = lds_f[r + 8 * hi];
        il2[r] = lds_f[16 + r + 8 * hi];
    }
#pragma unroll
    for (int nt = 0; nt < 8; ++nt) {
#pragma unroll
        for (int r = 0; r < 8; ++r) {
            const float o = (acc1[nt][r] * il1[r] - lam * acc2[nt][r] * il2[r]) * OUT_SCALE_C;
            out[(size_t)(q0 + r + 8 * hi) * (NH * DV) + h * DV + nt * 16 + lm] = (__bf16)o;
        }
    }
}

// ---------------------------------------------------------------------------
// Host-side launch
// ---------------------------------------------------------------------------
extern "C" void kernel_launch(void* const* d_in, const int* in_sizes, int n_in,
                              void* d_out, int out_size, void* d_ws, size_t ws_size,
                              hipStream_t stream)
{
    const float* x     = (const float*)d_in[0];   // [1,2048,2048]
    const float* wq_a  = (const float*)d_in[1];   // [768,2048]
    const float* wq_b  = (const float*)d_in[2];   // [3072,768]
    const float* wkv_a = (const float*)d_in[3];   // [576,2048]
    const float* wkv_b = (const float*)d_in[4];   // [4096,512]
    const float* wo    = (const float*)d_in[5];   // [2048,2048]
    const float* lqn   = (const float*)d_in[6];
    const float* lqr   = (const float*)d_in[7];
    const float* lkn   = (const float*)d_in[8];
    const float* lkr   = (const float*)d_in[9];
    const float* fcos  = (const float*)d_in[10];  // [2048,32]
    const float* fsin  = (const float*)d_in[11];  // [2048,32]
    float* outp        = (float*)d_out;           // [1,2048,2048] f32

    char* w = (char*)d_ws;
    size_t off = 0;
    auto take = [&](size_t bytes) -> char* {
        char* p = w + off;
        off = (off + bytes + 255) & ~(size_t)255;
        return p;
    };
    // bf16 copies of inputs
    __bf16* xb    = (__bf16*)take((size_t)SEQ * DIM * 2);
    __bf16* wqab  = (__bf16*)take((size_t)QLR * DIM * 2);
    __bf16* wqbb  = (__bf16*)take((size_t)NH * QKH * QLR * 2);
    __bf16* wkvab = (__bf16*)take((size_t)(KVLR + DR) * DIM * 2);
    __bf16* wkvbb = (__bf16*)take((size_t)NH * (DN + DV) * KVLR * 2);
    __bf16* wob   = (__bf16*)take((size_t)DIM * NH * DV * 2);
    // bf16 intermediates
    __bf16* q_lat   = (__bf16*)take((size_t)SEQ * QLR * 2);
    __bf16* qbuf    = (__bf16*)take((size_t)SEQ * NH * QKH * 2);     // = qf
    __bf16* kv_full = (__bf16*)take((size_t)SEQ * (KVLR + DR) * 2);
    __bf16* kvb     = (__bf16*)take((size_t)SEQ * NH * (DN + DV) * 2);
    __bf16* kfb     = (__bf16*)take((size_t)SEQ * NH * QKH * 2);
    __bf16* vtb     = (__bf16*)take((size_t)NH * DV * SEQ * 2);
    __bf16* attn_o  = (__bf16*)take((size_t)SEQ * NH * DV * 2);
    float*  lam_p   = (float*) take(256);

    auto cvt = [&](const float* src, __bf16* dst, size_t n) {
        cvt_bf16_kernel<<<(unsigned)((n / 4 + 255) / 256), 256, 0, stream>>>(src, dst, n);
    };
    cvt(x,     xb,    (size_t)SEQ * DIM);
    cvt(wq_a,  wqab,  (size_t)QLR * DIM);
    cvt(wq_b,  wqbb,  (size_t)NH * QKH * QLR);
    cvt(wkv_a, wkvab, (size_t)(KVLR + DR) * DIM);
    cvt(wkv_b, wkvbb, (size_t)NH * (DN + DV) * KVLR);
    cvt(wo,    wob,   (size_t)DIM * NH * DV);

    const dim3 gblk(128);
    // q_lat = x @ wq_a^T                (M=2048, K=2048, N=768)
    gemm_bt_kernel<true><<<dim3(QLR / 64, SEQ / 128), gblk, 0, stream>>>(
        xb, DIM, wqab, DIM, q_lat, QLR, DIM);
    // kv_full = x @ wkv_a^T             (N=576)
    gemm_bt_kernel<true><<<dim3((KVLR + DR) / 64, SEQ / 128), gblk, 0, stream>>>(
        xb, DIM, wkvab, DIM, kv_full, KVLR + DR, DIM);
    // q = q_lat @ wq_b^T                (K=768, N=3072)
    gemm_bt_kernel<true><<<dim3(NH * QKH / 64, SEQ / 128), gblk, 0, stream>>>(
        q_lat, QLR, wqbb, QLR, qbuf, NH * QKH, QLR);
    // kvb = kv_full[:, :512] @ wkv_b^T  (K=512, N=4096, lda=576)
    gemm_bt_kernel<true><<<dim3(NH * (DN + DV) / 64, SEQ / 128), gblk, 0, stream>>>(
        kv_full, KVLR + DR, wkvbb, KVLR, kvb, NH * (DN + DV), KVLR);

    lam_kernel<<<1, 32, 0, stream>>>(lqn, lqr, lkn, lkr, lam_p);

    prep_kernel<<<dim3(SEQ, NH), dim3(160), 0, stream>>>(
        qbuf, kvb, kv_full, fcos, fsin, kfb, vtb);

    diff_attn_kernel<<<dim3(SEQ / 16, NH), dim3(32), 0, stream>>>(
        qbuf, kfb, vtb, lam_p, attn_o);

    // out = attn_o @ wo^T               (M=2048, K=2048, N=2048, f32 out)
    gemm_bt_kernel<false><<<dim3(DIM / 64, SEQ / 128), gblk, 0, stream>>>(
        attn_o, NH * DV, wob, DIM, outp, DIM, DIM);
}
